// RNN_cls_st_11897059410483
// MI455X (gfx1250) — compile-verified, matrix operand order checked by reading, not run on hardware
//
#include <hip/hip_runtime.h>
#include <hip/hip_bf16.h>

// Problem constants (from reference)
#define S_   512
#define U_   128
#define H_   512
#define V_   2048
#define UH_  (U_*H_)
#define SUH_ ((size_t)S_*U_*H_)
#define DAY_  86400.0f
#define LAMT_ 0.1f
#define LAMS_ 1.0f

typedef __attribute__((ext_vector_type(16))) __bf16 v16bf;
typedef __attribute__((ext_vector_type(8)))  float  v8f;

struct Bar { unsigned count; unsigned phase; unsigned pad[14]; };

// ---------------------------------------------------------------------------
// WMMA helpers (CDNA5 wave32): D(16x16,f32) = A(16x32,bf16) x B(32x16,bf16) + C
// ---------------------------------------------------------------------------
__device__ __forceinline__ v8f wmma_bf16(v16bf a, v16bf b, v8f c) {
  return __builtin_amdgcn_wmma_f32_16x16x32_bf16(false, a, false, b, (short)0, c,
                                                 false, false);
}

// Fragment loader for sources stored contiguously along K (stride between
// rows/cols = `stride` elements). Works for both A (row m = lane&15) and
// B stored as [N][K] (col n = lane&15). Per ISA layout:
//   lanes 0-15 : K = {kb+0..kb+7, kb+16..kb+23} with kb = 0
//   lanes 16-31: same with kb = 8
__device__ __forceinline__ v16bf load_frag(const __bf16* base, int stride, int k0) {
  const int lane = threadIdx.x & 31;
  const int r    = lane & 15;
  const int kb   = (lane >> 4) << 3;
  const __bf16* p = base + (size_t)r * stride + k0 + kb;
  v16bf f;
#pragma unroll
  for (int e = 0; e < 8; ++e) f[e] = p[e];
#pragma unroll
  for (int e = 0; e < 8; ++e) f[8 + e] = p[16 + e];
  return f;
}

// ---------------------------------------------------------------------------
// Grid-wide barrier for the persistent GRU kernel (32 workgroups).
// ---------------------------------------------------------------------------
__device__ __forceinline__ void grid_barrier(Bar* b, unsigned nwg, unsigned* lp) {
  __syncthreads();
  if (threadIdx.x == 0) {
    const unsigned target = ++(*lp);
    const unsigned old = __hip_atomic_fetch_add(&b->count, 1u, __ATOMIC_ACQ_REL,
                                                __HIP_MEMORY_SCOPE_AGENT);
    if (old == nwg - 1u) {
      __hip_atomic_store(&b->count, 0u, __ATOMIC_RELAXED, __HIP_MEMORY_SCOPE_AGENT);
      __hip_atomic_store(&b->phase, target, __ATOMIC_RELEASE, __HIP_MEMORY_SCOPE_AGENT);
    } else {
      while (__hip_atomic_load(&b->phase, __ATOMIC_ACQUIRE,
                               __HIP_MEMORY_SCOPE_AGENT) < target)
        __builtin_amdgcn_s_sleep(1);
    }
  }
  __syncthreads();
}

__device__ __forceinline__ float sigmoidf_(float x) {
  return 1.0f / (1.0f + __expf(-x));
}

// ---------------------------------------------------------------------------
// Prep kernels
// ---------------------------------------------------------------------------
__global__ void __launch_bounds__(256) k_init(const float* __restrict__ h0,
                                              float* __restrict__ hF,
                                              __bf16* __restrict__ hB, Bar* bar) {
  if (blockIdx.x == 0 && threadIdx.x == 0) { bar->count = 0; bar->phase = 0; }
  for (int i = blockIdx.x * 256 + threadIdx.x; i < UH_; i += gridDim.x * 256) {
    const float v = h0[i];
    hF[i] = v;                 // f32 carry (buffer 0)
    hB[i] = (__bf16)v;         // bf16 GEMM feed (buffer 0)
  }
}

// WcatT[n][k], n in [0,1536), k in [0,1024): k<512 -> W_ih[n][k], else W_hh[n][k-512]
__global__ void __launch_bounds__(256) k_weights(const float* __restrict__ Wih,
                                                 const float* __restrict__ Whh,
                                                 const float* __restrict__ fcw,
                                                 __bf16* __restrict__ Wcat,
                                                 __bf16* __restrict__ fcwb) {
  const int idx = blockIdx.x * 256 + threadIdx.x;
  const int total1 = 3 * H_ * 1024;
  const int total2 = V_ * H_;
  for (int i = idx; i < total1; i += gridDim.x * 256) {
    const int n = i >> 10, k = i & 1023;
    const float v = (k < H_) ? Wih[n * H_ + k] : Whh[n * H_ + (k - H_)];
    Wcat[i] = (__bf16)v;
  }
  for (int i = idx; i < total2; i += gridDim.x * 256) fcwb[i] = (__bf16)fcw[i];
}

__global__ void __launch_bounds__(256) k_embed(const int* __restrict__ x,
                                               const float* __restrict__ emb,
                                               __bf16* __restrict__ Xbf) {
  for (size_t i = (size_t)blockIdx.x * 256 + threadIdx.x; i < SUH_;
       i += (size_t)gridDim.x * 256) {
    const int hcol = (int)(i & (H_ - 1));
    const int su   = (int)(i >> 9);         // H_ == 512
    const int tok  = x[su];
    Xbf[i] = (__bf16)emb[(size_t)tok * H_ + hcol];
  }
}

// ---------------------------------------------------------------------------
// Persistent GRU: 32 WGs x 8 waves = 256 waves, one 16x16 (u,h) tile each.
// Per step: [x_t ; h_prev](128x1024) @ WcatT(1024x1536) with split accumulators
// for the n-gate (inn vs hn), then fused gate math. Grid barrier between steps.
// K loop is peeled at K=512 so the hot loops are branch-free.
// ---------------------------------------------------------------------------
__global__ void __launch_bounds__(256) k_gru(const __bf16* __restrict__ Xbf,
                                             const __bf16* __restrict__ Wcat,
                                             const float* __restrict__ bih,
                                             const float* __restrict__ bhh,
                                             float* __restrict__ hF,
                                             __bf16* __restrict__ hB,
                                             __bf16* __restrict__ outb,
                                             float* __restrict__ hlast, Bar* bar) {
  const int lane = threadIdx.x & 31, wv = threadIdx.x >> 5;
  const int tile = blockIdx.x * 8 + wv;     // 0..255
  const int u0 = (tile >> 5) << 4;          // 8 user tiles
  const int h0 = (tile & 31) << 4;          // 32 hidden tiles
  const int ncol = lane & 15;
  const int mb   = (lane >> 4) << 3;

  // per-lane gate biases (column is fixed for the whole scan)
  const float br  = bih[h0 + ncol]          + bhh[h0 + ncol];
  const float bz  = bih[H_ + h0 + ncol]     + bhh[H_ + h0 + ncol];
  const float bin = bih[2 * H_ + h0 + ncol];
  const float bhn = bhh[2 * H_ + h0 + ncol];

  const __bf16* Br = Wcat + (size_t)(h0)          * 1024;
  const __bf16* Bz = Wcat + (size_t)(H_ + h0)     * 1024;
  const __bf16* Bn = Wcat + (size_t)(2 * H_ + h0) * 1024;

  unsigned phase = 0;
  for (int s = 0; s < S_; ++s) {
    const int rb = s & 1, wb = rb ^ 1;
    const __bf16* Xrow = Xbf + ((size_t)s * U_ + u0) * H_;
    const __bf16* Hrow = hB + (size_t)rb * UH_ + (size_t)u0 * H_;
    const float*  hFr  = hF + (size_t)rb * UH_;

    // Prefetch next step's X tile (16 rows x 1KB) into cache while we compute.
    if (s + 1 < S_) {
      const char* xn = (const char*)(Xbf + ((size_t)(s + 1) * U_ + u0) * H_);
      __builtin_prefetch(xn + (size_t)lane * 512, 0, 3);
    }

    v8f ar = {}, az = {}, ain = {}, ahn = {};
    // K in [0,512): input-gate half, A rows come from x_t
    for (int k0 = 0; k0 < H_; k0 += 32) {
      const v16bf a = load_frag(Xrow, H_, k0);
      ar  = wmma_bf16(a, load_frag(Br, 1024, k0), ar);
      az  = wmma_bf16(a, load_frag(Bz, 1024, k0), az);
      ain = wmma_bf16(a, load_frag(Bn, 1024, k0), ain);
    }
    // K in [512,1024): hidden-gate half, A rows come from h_{s-1}
    for (int k0 = H_; k0 < 1024; k0 += 32) {
      const v16bf a = load_frag(Hrow, H_, k0 - H_);
      ar  = wmma_bf16(a, load_frag(Br, 1024, k0), ar);
      az  = wmma_bf16(a, load_frag(Bz, 1024, k0), az);
      ahn = wmma_bf16(a, load_frag(Bn, 1024, k0), ahn);
    }

#pragma unroll
    for (int e = 0; e < 8; ++e) {
      const int u = u0 + mb + e;
      const float r  = sigmoidf_(ar[e] + br);
      const float z  = sigmoidf_(az[e] + bz);
      const float ng = tanhf((ain[e] + bin) + r * (ahn[e] + bhn));
      const float hp = hFr[(size_t)u * H_ + h0 + ncol];
      const float hn = (1.0f - z) * ng + z * hp;
      hF[(size_t)wb * UH_ + (size_t)u * H_ + h0 + ncol] = hn;
      const __bf16 hnb = (__bf16)hn;
      hB[(size_t)wb * UH_ + (size_t)u * H_ + h0 + ncol] = hnb;
      outb[((size_t)s * U_ + u) * H_ + h0 + ncol] = hnb;
    }
    grid_barrier(bar, gridDim.x, &phase);
  }

  // h_last: final state lives in f32 buffer 0 (step 511 wrote wb = 0);
  // each wave re-reads exactly the tile it wrote.
#pragma unroll
  for (int e = 0; e < 8; ++e) {
    const int u = u0 + mb + e;
    hlast[(size_t)u * H_ + h0 + ncol] = hF[(size_t)u * H_ + h0 + ncol];
  }
}

// ---------------------------------------------------------------------------
// Causal spatiotemporal attention: one WG per (user, 16-row i-tile).
// Build W tile (exp weights + row sums) in LDS, stage out[j-chunk] transposed
// in LDS, 8 waves x 4 accumulators cover all H columns via bf16 WMMA.
// ---------------------------------------------------------------------------
__global__ void __launch_bounds__(256) k_attn(const float* __restrict__ t,
                                              const float* __restrict__ sxy,
                                              const __bf16* __restrict__ outb,
                                              __bf16* __restrict__ outw) {
  __shared__ __bf16 outT[H_ * 34];   // transposed [h][j], padded stride 34
  __shared__ __bf16 wA[16 * 34];     // W tile [i][j]
  __shared__ float rowsum[16];
  __shared__ float tI[16], sI[32], tJ[32], sJ[64];

  const int u  = blockIdx.x >> 5;
  const int i0 = (blockIdx.x & 31) << 4;
  const int tid = threadIdx.x, lane = tid & 31, wv = tid >> 5;
  const int h0 = wv << 6;            // 64 H-columns per wave

  if (tid < 16) { tI[tid] = t[(i0 + tid) * U_ + u]; rowsum[tid] = 0.f; }
  if (tid < 32) sI[tid] = sxy[((size_t)(i0 + (tid >> 1)) * U_ + u) * 2 + (tid & 1)];

  v8f acc[4] = {{}, {}, {}, {}};
  const int nch = (i0 + 47) >> 5;    // j-chunks of 32 covering j <= i0+15
  for (int c = 0; c < nch; ++c) {
    const int j0 = c << 5;
    if (tid < 32) tJ[tid] = t[(j0 + tid) * U_ + u];
    if (tid < 64) sJ[tid] = sxy[((size_t)(j0 + (tid >> 1)) * U_ + u) * 2 + (tid & 1)];
    __syncthreads();

    // W tile: w = exp(-dt*LAMT/DAY - LAMS*|ds|) + 1e-10 for j<=i, else 0
    for (int r = tid; r < 512; r += 256) {
      const int ii = r >> 5, jj = r & 31;
      float val = 0.f;
      if (j0 + jj <= i0 + ii) {
        const float dt = tI[ii] - tJ[jj];
        const float dx = sI[2 * ii] - sJ[2 * jj];
        const float dy = sI[2 * ii + 1] - sJ[2 * jj + 1];
        val = __expf(-dt * (LAMT_ / DAY_) - LAMS_ * sqrtf(dx * dx + dy * dy)) + 1e-10f;
        atomicAdd(&rowsum[ii], val);
      }
      wA[ii * 34 + jj] = (__bf16)val;
    }
    // stage out[j0..j0+31, u, :] transposed so B frags are k-contiguous
    for (int r = tid; r < 32 * H_; r += 256) {
      const int hh = r & (H_ - 1), jj = r >> 9;
      outT[hh * 34 + jj] = outb[((size_t)(j0 + jj) * U_ + u) * H_ + hh];
    }
    __syncthreads();

    const v16bf a = load_frag(wA, 34, 0);
#pragma unroll
    for (int nt = 0; nt < 4; ++nt) {
      const v16bf b = load_frag(outT + (h0 + nt * 16) * 34, 34, 0);
      acc[nt] = wmma_bf16(a, b, acc[nt]);
    }
  }

  const int n = lane & 15, mb = (lane >> 4) << 3;
  float inv[8];
#pragma unroll
  for (int e = 0; e < 8; ++e) inv[e] = 1.0f / rowsum[mb + e];
#pragma unroll
  for (int nt = 0; nt < 4; ++nt) {
    const int hcol = h0 + nt * 16 + n;
#pragma unroll
    for (int e = 0; e < 8; ++e) {
      const int i = i0 + mb + e;
      outw[((size_t)i * U_ + u) * H_ + hcol] = (__bf16)(acc[nt][e] * inv[e]);
    }
  }
}

// ---------------------------------------------------------------------------
// Final FC: y[65536,2048] = out_w[65536,512] @ fc_w^T + fc_b   (bf16 WMMA)
// fc_w is [V,H] = [N,K] row-major -> k-contiguous B frags directly.
// ---------------------------------------------------------------------------
__global__ void __launch_bounds__(256) k_fc(const __bf16* __restrict__ outw,
                                            const __bf16* __restrict__ fcwb,
                                            const float* __restrict__ fcb,
                                            float* __restrict__ y) {
  const int lane = threadIdx.x & 31, wv = threadIdx.x >> 5;
  const size_t tile = (size_t)blockIdx.x * 8 + wv;
  const int n0 = (int)(tile & 127) << 4;        // V/16 = 128 tiles
  const int m0 = (int)(tile >> 7) << 4;         // (S*U)/16 = 4096 tiles
  const __bf16* A = outw + (size_t)m0 * H_;
  const __bf16* B = fcwb + (size_t)n0 * H_;
  v8f acc = {};
  for (int k0 = 0; k0 < H_; k0 += 32)
    acc = wmma_bf16(load_frag(A, H_, k0), load_frag(B, H_, k0), acc);
  const int n = lane & 15, mb = (lane >> 4) << 3;
  const float bias = fcb[n0 + n];
#pragma unroll
  for (int e = 0; e < 8; ++e)
    y[(size_t)(m0 + mb + e) * V_ + n0 + n] = acc[e] + bias;
}

// ---------------------------------------------------------------------------
extern "C" void kernel_launch(void* const* d_in, const int* in_sizes, int n_in,
                              void* d_out, int out_size, void* d_ws, size_t ws_size,
                              hipStream_t stream) {
  (void)in_sizes; (void)n_in; (void)out_size; (void)ws_size;
  const int*   x   = (const int*)  d_in[0];
  const float* t   = (const float*)d_in[1];
  const float* sxy = (const float*)d_in[2];
  const float* h0  = (const float*)d_in[5];
  const float* emb = (const float*)d_in[7];
  const float* Wih = (const float*)d_in[8];
  const float* Whh = (const float*)d_in[9];
  const float* bih = (const float*)d_in[10];
  const float* bhh = (const float*)d_in[11];
  const float* fcw = (const float*)d_in[12];
  const float* fcb = (const float*)d_in[13];

  float* y     = (float*)d_out;                     // [S,U,V]
  float* hlast = y + (size_t)S_ * U_ * V_;          // [1,U,H]

  char* w = (char*)d_ws;
  size_t off = 0;
  auto alloc = [&](size_t bytes) -> void* {
    void* p = w + off;
    off = (off + bytes + 255) & ~(size_t)255;
    return p;
  };
  Bar*    bar  = (Bar*)   alloc(sizeof(Bar));
  float*  hF   = (float*) alloc((size_t)2 * UH_ * sizeof(float));    // ping-pong f32
  __bf16* hB   = (__bf16*)alloc((size_t)2 * UH_ * sizeof(__bf16));   // ping-pong bf16
  __bf16* Wcat = (__bf16*)alloc((size_t)3 * H_ * 1024 * sizeof(__bf16));
  __bf16* fcwb = (__bf16*)alloc((size_t)V_ * H_ * sizeof(__bf16));
  __bf16* Xbf  = (__bf16*)alloc(SUH_ * sizeof(__bf16));
  __bf16* outb = (__bf16*)alloc(SUH_ * sizeof(__bf16));
  __bf16* outw = (__bf16*)alloc(SUH_ * sizeof(__bf16));

  k_init   <<<256,    256, 0, stream>>>(h0, hF, hB, bar);
  k_weights<<<4096,   256, 0, stream>>>(Wih, Whh, fcw, Wcat, fcwb);
  k_embed  <<<32768,  256, 0, stream>>>(x, emb, Xbf);
  k_gru    <<<32,     256, 0, stream>>>(Xbf, Wcat, bih, bhh, hF, hB, outb, hlast, bar);
  k_attn   <<<U_ * 32,256, 0, stream>>>(t, sxy, outb, outw);
  k_fc     <<<65536,  256, 0, stream>>>(outw, fcwb, fcb, y);
}
